// CNN_7009386627340
// MI455X (gfx1250) — 8-bit matrix operand order repaired and checked by reading, compiled, not run on hardware
//
#include <hip/hip_runtime.h>
#include <hip/hip_bf16.h>


typedef int   v8i __attribute__((ext_vector_type(8)));
typedef float v8f __attribute__((ext_vector_type(8)));
typedef float v2f __attribute__((ext_vector_type(2)));

#define NB      8192
#define C1      16
#define C2      32
#define HW1     28
#define P1      14
#define Q2      7
#define K1CH    8
#define K2CH    8
#define FCK     1568
#define FCCH    25
#define BANDS1  (NB * 14)
#define NBLK1   (BANDS1 / 4)
#define BANDS2  (NB * 7)
#define NBLK2   (BANDS2 / 4)
#define FCTILE  (NB / 16)
#define EPSBN   1e-6

#define OFF_PARAMS   0u
#define OFF_PB2      512u
#define OFF_PBFC     16896u
#define OFF_PART1    42496u
#define OFF_PART2    3712512u
#define OFF_A1       7382528u
#define OFF_A2       33072640u

#define IMGW 40
#define IMGR 13
__host__ __device__ constexpr int oOff1(int kk, int vi) {
    int s = kk * 2 + vi;
    return (s < 15) ? (s / 3) * IMGW + 2 * (s % 3) : 6 * IMGW;
}

#define CIBS 240
__host__ __device__ constexpr int oOff2(int v, int j) {
    int op = ((v & 4) ? 16 : 0) + (((v >> 1) & 1) << 3) + ((v & 1) << 2) + j;
    return (op < 25) ? (op / 5) * 20 + (op % 5) : 120;
}
__device__ __forceinline__ int kposA(int v, int half) {
    return ((v >= 4) ? 32 : 0) + half * 16 + (v & 3) * 4;
}
__host__ __device__ __forceinline__ int kposB(int v, int half) {
    return ((v >= 4) ? 32 : 0) + half * 16 + (v & 3) * 4;
}

__global__ void pack_weights(const float* __restrict__ w2, const float* __restrict__ fcw,
                             unsigned* __restrict__ pb2, unsigned* __restrict__ pbfc) {
    int t = blockIdx.x * blockDim.x + threadIdx.x;
    if (t < K2CH * 2 * 32 * 8) {
        int v = t & 7, L = (t >> 3) & 31, nt = (t >> 8) & 1, c = t >> 9;
        int n = L & 15, halfL = L >> 4;
        int co = nt * 16 + n;
        unsigned u = 0;
        for (int j = 0; j < 4; ++j) {
            int k = 64 * c + kposB(v, halfL) + j;
            int ci = 2 * (k >> 6) + ((k >> 4) & 1);
            int op = ((k >> 5) & 1) * 16 + ((k >> 3) & 1) * 8 + ((k >> 2) & 1) * 4 + (k & 3);
            int byte = 0;
            if (op < 25) byte = (w2[co * 400 + ci * 25 + op] >= 0.f) ? 1 : -1;
            u |= ((unsigned)(unsigned char)(signed char)byte) << (8 * j);
        }
        pb2[t] = u;
    } else {
        int s = t - K2CH * 2 * 32 * 8;
        if (s < FCCH * 32 * 8) {
            int v = s & 7, L = (s >> 3) & 31, c = s >> 8;
            int n = L & 15, halfL = L >> 4;
            unsigned u = 0;
            for (int j = 0; j < 4; ++j) {
                int k = 64 * c + kposB(v, halfL) + j;
                int byte = 0;
                if (n < 10 && k < FCK) byte = (fcw[n * FCK + k] >= 0.f) ? 1 : -1;
                u |= ((unsigned)(unsigned char)(signed char)byte) << (8 * j);
            }
            pbfc[s] = u;
        }
    }
}

__global__ void conv1_wmma(const float* __restrict__ x, const float* __restrict__ w1,
                           const float* __restrict__ b1, const float* __restrict__ params,
                           float* __restrict__ part1, signed char* __restrict__ a1, int mode) {
    __shared__ float img[4][IMGR * IMGW];
    __shared__ float ylds[4][64][16];
    __shared__ float part[4][16][2];
    int wave = threadIdx.x >> 5, lane = threadIdx.x & 31;
    int band = blockIdx.x * 4 + wave;
    int b = band / 14, p = band % 14;
    int n = lane & 15, half = lane >> 4;

    for (int i = lane; i < IMGR * IMGW; i += 32) {
        int r = i / IMGW, cc = i - r * IMGW;
        float v = 0.f;
        int iy = 2 * p - 2 + r, ix = cc - 2;
        if (r < 6 && (unsigned)iy < (unsigned)HW1 && (unsigned)ix < (unsigned)HW1)
            v = x[(size_t)b * (HW1 * HW1) + iy * HW1 + ix];
        img[wave][i] = v;
    }
    v2f Bv[K1CH];
#pragma unroll
    for (int kk = 0; kk < K1CH; ++kk)
#pragma unroll
        for (int vi = 0; vi < 2; ++vi) {
            int s = kk * 2 + vi;
            int ky = s / 3, kx = 2 * (s % 3) + half;
            float bw = 0.f;
            if (s < 15 && kx < 5) bw = (w1[n * 25 + ky * 5 + kx] >= 0.f) ? 1.f : -1.f;
            Bv[kk][vi] = bw;
        }
    __syncthreads();

    float bias = b1[n];
    float sc = 1.f, sh = 0.f;
    if (mode == 1) { sc = params[n]; sh = params[16 + n]; }

#pragma unroll
    for (int mt = 0; mt < 4; ++mt) {
        int sp = mt * 16 + n;
        int dy = (sp >= 28) ? 1 : 0, ox = sp - 28 * dy;
        const float* pa = &img[wave][0] + ((sp < 56) ? dy * IMGW + ox : 6 * IMGW) + half;
        v8f acc = {};
#pragma unroll
        for (int kk = 0; kk < K1CH; ++kk) {
            v2f A;
            A[0] = pa[oOff1(kk, 0)];
            A[1] = pa[oOff1(kk, 1)];
            acc = __builtin_amdgcn_wmma_f32_16x16x4_f32(false, A, false, Bv[kk],
                                                        (short)0, acc, false, false);
        }
#pragma unroll
        for (int r = 0; r < 8; ++r) {
            int sg = mt * 16 + r + half * 8;
            float y = acc[r] + bias;
            if (mode == 1) y = y * sc + sh;
            else if (sg >= 56) y = 0.f;
            ylds[wave][sg][n] = y;
        }
    }
    __syncthreads();

    if (mode == 0) {
        int ch = lane & 15, st = lane >> 4;
        float a = 0.f;
        for (int s = 0; s < 56; ++s) {
            float y = ylds[wave][s][ch];
            a += st ? y * y : y;
        }
        part[wave][ch][st] = a;
        __syncthreads();
        if (threadIdx.x < 32) {
            int ch2 = threadIdx.x & 15, st2 = threadIdx.x >> 4;
            float S = 0.f;
#pragma unroll
            for (int w = 0; w < 4; ++w) S += part[w][ch2][st2];
            part1[((size_t)blockIdx.x * 16 + ch2) * 2 + st2] = S;
        }
    } else {
        int ch = lane & 15;
#pragma unroll
        for (int i = 0; i < 7; ++i) {
            int px = (lane >> 4) * 7 + i;
            float v0 = ylds[wave][2 * px][ch],      v1 = ylds[wave][2 * px + 1][ch];
            float v2 = ylds[wave][28 + 2 * px][ch], v3 = ylds[wave][29 + 2 * px][ch];
            float m = fmaxf(fmaxf(v0, v1), fmaxf(v2, v3));
            a1[(((size_t)b * C1 + ch) * P1 + p) * P1 + px] = (m >= 0.f) ? 1 : -1;
        }
    }
}

__global__ void stats1_final(const float* __restrict__ part1, const float* __restrict__ g,
                             const float* __restrict__ beta, float* __restrict__ params) {
    int c = threadIdx.x;
    if (c >= C1) return;
    double s = 0.0, q = 0.0;
    for (int blk = 0; blk < NBLK1; ++blk) {
        s += part1[((size_t)blk * 16 + c) * 2 + 0];
        q += part1[((size_t)blk * 16 + c) * 2 + 1];
    }
    double N = (double)NB * (HW1 * HW1);
    double mean = s / N, var = q / N - mean * mean;
    double inv = 1.0 / sqrt(var + EPSBN);
    float scale = (float)((double)g[c] * inv);
    params[c] = scale;
    params[16 + c] = (float)((double)beta[c] - mean * (double)scale);
}

__global__ void conv2_wmma(const signed char* __restrict__ a1g, const unsigned* __restrict__ pb2,
                           const float* __restrict__ b2, const float* __restrict__ params,
                           float* __restrict__ part2, signed char* __restrict__ a2, int mode) {
    __shared__ unsigned char aLds[4][16 * CIBS];
    __shared__ float ylds[4][32][32];
    __shared__ float part[4][32][2];
    int wave = threadIdx.x >> 5, lane = threadIdx.x & 31;
    int band = blockIdx.x * 4 + wave;
    int b = band / 7, q = band % 7;
    int n = lane & 15, half = lane >> 4;

    unsigned* zp = (unsigned*)&aLds[wave][0];
    for (int i = lane; i < (16 * CIBS) / 4; i += 32) zp[i] = 0;
    __syncthreads();
#pragma unroll
    for (int it = 0; it < 3; ++it) {
        int t = lane + it * 32;
        int ci = t / 6, r = t - ci * 6;
        int iy = 2 * q - 2 + r;
        if ((unsigned)iy < (unsigned)P1) {
            const unsigned short* src =
                (const unsigned short*)(a1g + (((size_t)b * C1 + ci) * P1 + iy) * P1);
            unsigned short* dst = (unsigned short*)&aLds[wave][ci * CIBS + r * 20 + 2];
#pragma unroll
            for (int i = 0; i < 7; ++i) dst[i] = src[i];
        }
    }
    __syncthreads();

    int pix0 = (n >= 14) ? (20 + n - 14) : n;
    int pix1 = (n < 12) ? (22 + n) : 120;
    const unsigned char* pa0 = &aLds[wave][0] + half * CIBS + pix0;
    const unsigned char* pa1 = &aLds[wave][0] + half * CIBS + pix1;
    const unsigned* pbl = pb2 + lane * 8;

    v8i Acc[2][2] = {};
    for (int c = 0; c < K2CH; ++c) {
        v8i A0, A1, B0, B1;
#pragma unroll
        for (int v = 0; v < 8; ++v) {
            unsigned u0 = 0, u1 = 0;
#pragma unroll
            for (int j = 0; j < 4; ++j) {
                u0 |= ((unsigned)pa0[oOff2(v, j)]) << (8 * j);
                u1 |= ((unsigned)pa1[oOff2(v, j)]) << (8 * j);
            }
            A0[v] = (int)u0; A1[v] = (int)u1;
            B0[v] = (int)pbl[(c * 2 + 0) * 256 + v];
            B1[v] = (int)pbl[(c * 2 + 1) * 256 + v];
        }
        Acc[0][0] = __builtin_amdgcn_wmma_i32_16x16x64_iu8(true, A0, true, B0, Acc[0][0], false, false);
        Acc[0][1] = __builtin_amdgcn_wmma_i32_16x16x64_iu8(true, A0, true, B1, Acc[0][1], false, false);
        Acc[1][0] = __builtin_amdgcn_wmma_i32_16x16x64_iu8(true, A1, true, B0, Acc[1][0], false, false);
        Acc[1][1] = __builtin_amdgcn_wmma_i32_16x16x64_iu8(true, A1, true, B1, Acc[1][1], false, false);
        pa0 += 2 * CIBS;
        pa1 += 2 * CIBS;
    }

#pragma unroll
    for (int mt = 0; mt < 2; ++mt)
#pragma unroll
        for (int nt = 0; nt < 2; ++nt) {
            int co = nt * 16 + n;
            float bias = b2[co];
            float s2 = 1.f, h2 = 0.f;
            if (mode == 1) { s2 = params[32 + co]; h2 = params[64 + co]; }
#pragma unroll
            for (int r = 0; r < 8; ++r) {
                int mg = mt * 16 + r + half * 8;
                float y = (float)Acc[mt][nt][r] + bias;
                if (mode == 1) y = y * s2 + h2;
                else if (mg >= 28) y = 0.f;
                ylds[wave][mg][co] = y;
            }
        }
    __syncthreads();

    if (mode == 0) {
        int ch = lane;
        float s = 0.f, qq = 0.f;
        for (int m = 0; m < 28; ++m) { float y = ylds[wave][m][ch]; s += y; qq += y * y; }
        part[wave][ch][0] = s; part[wave][ch][1] = qq;
        __syncthreads();
        if (threadIdx.x < 32) {
            float S = 0.f, Qq = 0.f;
#pragma unroll
            for (int w = 0; w < 4; ++w) { S += part[w][threadIdx.x][0]; Qq += part[w][threadIdx.x][1]; }
            part2[((size_t)blockIdx.x * 32 + threadIdx.x) * 2 + 0] = S;
            part2[((size_t)blockIdx.x * 32 + threadIdx.x) * 2 + 1] = Qq;
        }
    } else {
        int ch = lane;
#pragma unroll
        for (int px = 0; px < Q2; ++px) {
            float v0 = ylds[wave][2 * px][ch],      v1 = ylds[wave][2 * px + 1][ch];
            float v2 = ylds[wave][14 + 2 * px][ch], v3 = ylds[wave][15 + 2 * px][ch];
            float m = fmaxf(fmaxf(v0, v1), fmaxf(v2, v3));
            a2[(((size_t)b * C2 + ch) * Q2 + q) * Q2 + px] = (m >= 0.f) ? 1 : -1;
        }
    }
}

__global__ void stats2_final(const float* __restrict__ part2, const float* __restrict__ g,
                             const float* __restrict__ beta, float* __restrict__ params) {
    int c = threadIdx.x;
    if (c >= C2) return;
    double s = 0.0, q = 0.0;
    for (int blk = 0; blk < NBLK2; ++blk) {
        s += part2[((size_t)blk * 32 + c) * 2 + 0];
        q += part2[((size_t)blk * 32 + c) * 2 + 1];
    }
    double N = (double)NB * (P1 * P1);
    double mean = s / N, var = q / N - mean * mean;
    double inv = 1.0 / sqrt(var + EPSBN);
    float scale = (float)((double)g[c] * inv);
    params[32 + c] = scale;
    params[64 + c] = (float)((double)beta[c] - mean * (double)scale);
}

__global__ void fc_wmma(const signed char* __restrict__ a2, const unsigned* __restrict__ pbfc,
                        const float* __restrict__ fcb, float* __restrict__ out) {
    int wave = threadIdx.x >> 5, lane = threadIdx.x & 31;
    int tile = blockIdx.x * 4 + wave;
    int n = lane & 15, half = lane >> 4;

    v8i Acc = {};
    for (int c = 0; c < FCCH; ++c) {
        v8i A, Bv;
        int row = tile * 16 + n;
#pragma unroll
        for (int v = 0; v < 8; ++v) {
            int kbase = 64 * c + kposA(v, half);
            unsigned u = 0;
            if (kbase < FCK)
                u = *(const unsigned*)(a2 + (size_t)row * FCK + kbase);
            A[v] = (int)u;
            Bv[v] = (int)pbfc[((size_t)c * 32 + lane) * 8 + v];
        }
        Acc = __builtin_amdgcn_wmma_i32_16x16x64_iu8(true, A, true, Bv, Acc, false, false);
    }
    if (n < 10) {
        float bias = fcb[n];
#pragma unroll
        for (int r = 0; r < 8; ++r) {
            int bg = tile * 16 + r + half * 8;
            out[(size_t)bg * 10 + n] = (float)Acc[r] + bias;
        }
    }
}

extern "C" void kernel_launch(void* const* d_in, const int* in_sizes, int n_in,
                              void* d_out, int out_size, void* d_ws, size_t ws_size,
                              hipStream_t stream) {
    const float* x   = (const float*)d_in[0];
    const float* w1  = (const float*)d_in[1];
    const float* b1  = (const float*)d_in[2];
    const float* g1  = (const float*)d_in[3];
    const float* be1 = (const float*)d_in[4];
    const float* w2  = (const float*)d_in[5];
    const float* b2  = (const float*)d_in[6];
    const float* g2  = (const float*)d_in[7];
    const float* be2 = (const float*)d_in[8];
    const float* fcw = (const float*)d_in[9];
    const float* fcb = (const float*)d_in[10];
    float* out = (float*)d_out;

    char* ws = (char*)d_ws;
    float*       params = (float*)(ws + OFF_PARAMS);
    unsigned*    pb2    = (unsigned*)(ws + OFF_PB2);
    unsigned*    pbfc   = (unsigned*)(ws + OFF_PBFC);
    float*       part1  = (float*)(ws + OFF_PART1);
    float*       part2  = (float*)(ws + OFF_PART2);
    signed char* a1     = (signed char*)(ws + OFF_A1);
    signed char* a2     = (signed char*)(ws + OFF_A2);

    pack_weights<<<41, 256, 0, stream>>>(w2, fcw, pb2, pbfc);
    conv1_wmma<<<NBLK1, 128, 0, stream>>>(x, w1, b1, params, part1, a1, 0);
    stats1_final<<<1, 32, 0, stream>>>(part1, g1, be1, params);
    conv1_wmma<<<NBLK1, 128, 0, stream>>>(x, w1, b1, params, part1, a1, 1);
    conv2_wmma<<<NBLK2, 128, 0, stream>>>(a1, pb2, b2, params, part2, a2, 0);
    stats2_final<<<1, 32, 0, stream>>>(part2, g2, be2, params);
    conv2_wmma<<<NBLK2, 128, 0, stream>>>(a1, pb2, b2, params, part2, a2, 1);
    fc_wmma<<<FCTILE / 4, 128, 0, stream>>>(a2, pbfc, fcb, out);
}
